// PCGrimAge_53936199303238
// MI455X (gfx1250) — matrix operand order repaired and checked by reading, not compiled, not run on hardware
//
#include <hip/hip_runtime.h>
#include <hip/hip_bf16.h>
#include <stddef.h>
#include <stdint.h>

// ---------------------------------------------------------------------------
// PCGrimAge pipeline for MI455X (gfx1250, wave32, WMMA, async LDS copies).
// Kernel 1: p[256,1936] = (x[:, :78464] - center) @ rotation  via bf16 WMMA
//           - triple-buffered LDS, GLOBAL_LOAD_ASYNC_TO_LDS staging,
//             software pipelined with s_wait_asynccnt.
// Kernel 2: gather/weighted-sum head -> out[256]
// ---------------------------------------------------------------------------

typedef __attribute__((ext_vector_type(16))) __bf16 v16bf;
typedef __attribute__((ext_vector_type(8)))  float  v8f;
typedef __attribute__((ext_vector_type(2)))  int    v2i_t;
typedef __attribute__((ext_vector_type(4)))  int    v4i_t;

typedef __attribute__((address_space(1))) v2i_t g_v2i;
typedef __attribute__((address_space(1))) v4i_t g_v4i;
typedef __attribute__((address_space(3))) v2i_t l_v2i;
typedef __attribute__((address_space(3))) v4i_t l_v4i;

#define B_ROWS   256
#define IN_DIM   78464
#define X_STRIDE 78466
#define PC_DIM   1936
#define KC       9
#define CC       600

#define BM 128          // WG tile rows
#define BN 64           // WG tile cols
#define BK 32           // K step
#define LDA 36          // padded LDS row stride (floats) for A tile
#define LDB 68          // padded LDS row stride (floats) for B tile
#define NSTAGE 3        // triple buffer (issue-ahead pipeline)

// --- CDNA5 async global->LDS copies (guarded; sync fallback keeps compile) ---
#if defined(__has_builtin)
#  if __has_builtin(__builtin_amdgcn_global_load_async_to_lds_b64) && \
      __has_builtin(__builtin_amdgcn_global_load_async_to_lds_b128)
#    define HAVE_ASYNC 1
#  endif
#  if __has_builtin(__builtin_amdgcn_s_wait_asynccnt)
#    define HAVE_WAIT_ASYNC 1
#  endif
#endif

__device__ __forceinline__ void cp_b64(float* lds, const float* g) {
#ifdef HAVE_ASYNC
    __builtin_amdgcn_global_load_async_to_lds_b64(
        (g_v2i*)(v2i_t*)(void*)(float*)g,
        (l_v2i*)(v2i_t*)(void*)lds, 0, 0);
#else
    *(float2*)lds = *(const float2*)g;
#endif
}

__device__ __forceinline__ void cp_b128(float* lds, const float* g) {
#ifdef HAVE_ASYNC
    __builtin_amdgcn_global_load_async_to_lds_b128(
        (g_v4i*)(v4i_t*)(void*)(float*)g,
        (l_v4i*)(v4i_t*)(void*)lds, 0, 0);
#else
    *(float4*)lds = *(const float4*)g;
#endif
}

#ifdef HAVE_ASYNC
#  ifdef HAVE_WAIT_ASYNC
#    define WAIT_ASYNC(n) __builtin_amdgcn_s_wait_asynccnt(n)
#  else
#    define WAIT_ASYNC(n) asm volatile("s_wait_asynccnt " #n ::: "memory")
#  endif
#else
#  define WAIT_ASYNC(n) do {} while (0)
#endif

__global__ __launch_bounds__(256) void pc_gemm_kernel(
    const float* __restrict__ x,
    const float* __restrict__ center,
    const float* __restrict__ rot,
    float* __restrict__ p)
{
    __shared__ float As[NSTAGE][BM][LDA];   // 55.3 KB
    __shared__ float Bs[NSTAGE][BK][LDB];   // 26.1 KB
    __shared__ float Cs[NSTAGE][BK];        // center chunks

    const int t    = threadIdx.x;
    const int lane = t & 31;
    const int wave = t >> 5;           // 0..7
    const int wm   = wave & 3;         // M group: rows wm*32 .. wm*32+31
    const int wn   = wave >> 2;        // N group: cols wn*32 .. wn*32+31

    const int row0 = blockIdx.x * BM;
    const int col0 = blockIdx.y * BN;

    v8f acc[2][2];
    const v8f vzero = {0.f, 0.f, 0.f, 0.f, 0.f, 0.f, 0.f, 0.f};
    acc[0][0] = vzero; acc[0][1] = vzero; acc[1][0] = vzero; acc[1][1] = vzero;

    // Fragment lane decode (ISA 7.12.2, 16-bit A 16x32 / B 32x16 layouts)
    const int am  = lane & 15;
    const int akb = (lane >> 4) << 3;    // A: K base 0 or 8
    const int bn  = lane & 15;
    const int bkb = (lane >> 4) << 3;    // B: K base 0 or 8

    const int ksteps = IN_DIM / BK;      // 2452

    // Per-tile stage: 8x async b64 (A) + 2x async b128 (B) = 10 asyncs/wave.
    auto stage_tile = [&](int sb, int kt) {
        const int k0 = kt * BK;
        #pragma unroll
        for (int j = 0; j < 8; ++j) {
            const int idx = t + j * 256;           // 0..2047
            const int r   = idx >> 4;              // 0..127
            const int c   = (idx & 15) * 2;        // 0..30 step 2 (8B aligned)
            cp_b64(&As[sb][r][c],
                   x + (size_t)(row0 + r) * X_STRIDE + k0 + c);
        }
        #pragma unroll
        for (int j = 0; j < 2; ++j) {
            const int idx = t + j * 256;           // 0..511
            const int r   = idx >> 4;              // 0..31
            const int c   = (idx & 15) * 4;        // 0..60 step 4 (16B aligned)
            if (col0 + c < PC_DIM)                 // OOB cols: stale LDS is
                cp_b128(&Bs[sb][r][c],             // masked at the epilogue
                        rot + (size_t)(k0 + r) * PC_DIM + col0 + c);
        }
        if (t < 8)
            *(float4*)(&Cs[sb][t * 4]) = *(const float4*)(center + k0 + t * 4);
    };

    stage_tile(0, 0);                    // prologue: tile 0 in flight

    int buf = 0, nbuf = 1;
    for (int ks = 0; ks < ksteps; ++ks) {
        if (ks + 1 < ksteps) {
            stage_tile(nbuf, ks + 1);    // issue ahead: tile ks+1
            // GL2 warm-up two tiles ahead for rotation (the big stream)
            if (ks + 2 < ksteps) {
                const int pr = t >> 4, pc = (t & 15) * 4;
                if (col0 + pc < PC_DIM)
                    __builtin_prefetch(
                        rot + (size_t)((ks + 2) * BK + pr) * PC_DIM + col0 + pc, 0, 0);
            }
            WAIT_ASYNC(10);              // tile ks complete (in-order ASYNCcnt)
        } else {
            WAIT_ASYNC(0);
        }
        __syncthreads();                 // one barrier/iter; NSTAGE=3 covers WAR

        // ---- build bf16 fragments from LDS, subtract center in f32 ----
        v16bf aF[2], bF[2];
        #pragma unroll
        for (int ms = 0; ms < 2; ++ms) {
            const float* arow = &As[buf][wm * 32 + ms * 16 + am][0];
            #pragma unroll
            for (int i = 0; i < 4; ++i) {
                const float2 lo = *(const float2*)(arow + akb + 2 * i);
                const float2 hi = *(const float2*)(arow + akb + 16 + 2 * i);
                aF[ms][2 * i]         = (__bf16)(lo.x - Cs[buf][akb + 2 * i]);
                aF[ms][2 * i + 1]     = (__bf16)(lo.y - Cs[buf][akb + 2 * i + 1]);
                aF[ms][8 + 2 * i]     = (__bf16)(hi.x - Cs[buf][akb + 16 + 2 * i]);
                aF[ms][8 + 2 * i + 1] = (__bf16)(hi.y - Cs[buf][akb + 16 + 2 * i + 1]);
            }
        }
        #pragma unroll
        for (int ns = 0; ns < 2; ++ns) {
            const int ccol = wn * 32 + ns * 16 + bn;
            #pragma unroll
            for (int i = 0; i < 8; ++i) {
                bF[ns][2 * i]     = (__bf16)Bs[buf][bkb + 2 * i + (i >= 4 ? 8 : 0)][ccol];
                bF[ns][2 * i + 1] = (__bf16)Bs[buf][bkb + 2 * i + 1 + (i >= 4 ? 8 : 0)][ccol];
            }
        }

        // ---- 4 WMMAs per wave per K step ----
        #pragma unroll
        for (int ms = 0; ms < 2; ++ms)
            #pragma unroll
            for (int ns = 0; ns < 2; ++ns)
                acc[ms][ns] = __builtin_amdgcn_wmma_f32_16x16x32_bf16(
                    false, aF[ms], false, bF[ns],
                    (short)0, acc[ms][ns], false, false);

        buf = nbuf;
        nbuf = (nbuf + 1 == NSTAGE) ? 0 : nbuf + 1;
    }

    // ---- epilogue: C/D layout (vgpr r => M=r(+8), lane&15 => N) ----
    #pragma unroll
    for (int ms = 0; ms < 2; ++ms) {
        #pragma unroll
        for (int ns = 0; ns < 2; ++ns) {
            const int gcol = col0 + wn * 32 + ns * 16 + (lane & 15);
            if (gcol < PC_DIM) {
                const int rbase = row0 + wm * 32 + ms * 16 + ((lane >> 4) << 3);
                #pragma unroll
                for (int r = 0; r < 8; ++r)
                    p[(size_t)(rbase + r) * PC_DIM + gcol] = acc[ms][ns][r];
            }
        }
    }
}

// ---------------------------------------------------------------------------
// Head: s_k = sum_c p_ext[b, feats[k,c]]*w1[k,c] + b1[k]
//       out[b] = sum_k s_k*w2[k] + age*w2[9] + female*w2[10] + b2
// p_ext = [p(1936), female, age]
// ---------------------------------------------------------------------------
__global__ __launch_bounds__(128) void pc_head_kernel(
    const float* __restrict__ p,
    const float* __restrict__ x,
    const int*   __restrict__ feats,
    const float* __restrict__ w1,
    const float* __restrict__ b1,
    const float* __restrict__ w2,
    const float* __restrict__ b2,
    float* __restrict__ out)
{
    __shared__ float pe[PC_DIM + 2];
    __shared__ float red[128];

    const int b = blockIdx.x;
    const int t = threadIdx.x;

    const float female = x[(size_t)b * X_STRIDE + IN_DIM];
    const float age    = x[(size_t)b * X_STRIDE + IN_DIM + 1];

    for (int i = t; i < PC_DIM; i += 128)
        pe[i] = p[(size_t)b * PC_DIM + i];
    if (t == 0) { pe[PC_DIM] = female; pe[PC_DIM + 1] = age; }
    __syncthreads();

    float partial = 0.f;
    for (int i = t; i < KC * CC; i += 128) {
        const int k   = i / CC;
        const int c   = i - k * CC;
        const int idx = feats[k * CC + c];
        partial += w2[k] * w1[k * CC + c] * pe[idx];
    }
    red[t] = partial;
    __syncthreads();
    #pragma unroll
    for (int s = 64; s > 0; s >>= 1) {
        if (t < s) red[t] += red[t + s];
        __syncthreads();
    }
    if (t == 0) {
        float r = red[0];
        #pragma unroll
        for (int k = 0; k < KC; ++k) r += w2[k] * b1[k];
        r += w2[KC] * age + w2[KC + 1] * female + b2[0];
        out[b] = r;
    }
}

extern "C" void kernel_launch(void* const* d_in, const int* in_sizes, int n_in,
                              void* d_out, int out_size, void* d_ws, size_t ws_size,
                              hipStream_t stream)
{
    const float* x      = (const float*)d_in[0];
    const float* center = (const float*)d_in[1];
    const float* rot    = (const float*)d_in[2];
    const int*   feats  = (const int*)d_in[3];
    const float* w1     = (const float*)d_in[4];
    const float* b1     = (const float*)d_in[5];
    const float* w2     = (const float*)d_in[6];
    const float* b2     = (const float*)d_in[7];
    float* out = (float*)d_out;
    float* p   = (float*)d_ws;     // 256*1936*4 = 1.98 MB scratch

    dim3 grid(B_ROWS / BM, (PC_DIM + BN - 1) / BN);   // (2, 31)
    pc_gemm_kernel<<<grid, 256, 0, stream>>>(x, center, rot, p);
    pc_head_kernel<<<B_ROWS, 128, 0, stream>>>(p, x, feats, w1, b1, w2, b2, out);
}
